// LSTMNetwork_31671088840825
// MI455X (gfx1250) — matrix operand order repaired and checked by reading, not compiled, not run on hardware
//
#include <hip/hip_runtime.h>
#include <stdint.h>
#include <math.h>

typedef __attribute__((ext_vector_type(16))) _Float16 v16h;
typedef __attribute__((ext_vector_type(8)))  _Float16 v8h;
typedef __attribute__((ext_vector_type(8)))  float    v8f;
typedef int v4i_ __attribute__((vector_size(16)));     // matches builtin's pointee

#define B_  32
#define T_  2048
#define I_  512
#define H_  512
#define O_  256
#define M_  (B_ * T_)      /* 65536 */
#define N_  (4 * H_)       /* 2048: gates i,f,g,o concatenated along N */
#define K_  I_             /* 512 */

#define TM  64
#define TN  128
#define TK  32
#define SXS 40             /* padded LDS row stride (halves): 80B, conflict-free */
#define SWS 40
#define NKS (K_ / TK)      /* 16 K-steps */

// ---------------------------------------------------------------------------
// CDNA5 async global->LDS copy (ASYNCcnt path).
// Signature (from hipcc diagnostic): param0 = AS1 (global) v4i*, non-const.
// ---------------------------------------------------------------------------
#if defined(__HIP_DEVICE_COMPILE__) && __has_builtin(__builtin_amdgcn_global_load_async_to_lds_b128)
#define CP_ASYNC_BUILTIN 1
#endif

typedef __attribute__((address_space(1))) v4i_ gas_v4i;
typedef __attribute__((address_space(3))) v4i_ las_v4i;

__device__ __forceinline__ void cp_async_b128(void* lds, const void* glb) {
#if defined(CP_ASYNC_BUILTIN)
    __builtin_amdgcn_global_load_async_to_lds_b128(
        (gas_v4i*)(unsigned long long)(uintptr_t)glb,
        (las_v4i*)(unsigned)(uintptr_t)lds,
        0, 0);
#else
    unsigned loff = (unsigned)(uintptr_t)lds;   // low 32 bits of generic ptr = LDS offset
    asm volatile("global_load_async_to_lds_b128 %0, %1, off"
                 :: "v"(loff), "v"((unsigned long long)(uintptr_t)glb) : "memory");
#endif
}

__device__ __forceinline__ void wait_async0() {
#if defined(__HIP_DEVICE_COMPILE__) && __has_builtin(__builtin_amdgcn_s_wait_asynccnt)
    __builtin_amdgcn_s_wait_asynccnt(0);
#else
    asm volatile("s_wait_asynccnt 0x0" ::: "memory");
#endif
}

// ---------------------------------------------------------------------------
// Fast activations: v_exp_f32 + v_rcp_f32, no IEEE-div / libm sequences.
// ---------------------------------------------------------------------------
__device__ __forceinline__ float fast_sigmoid(float v) {
    float e = __builtin_amdgcn_exp2f(-1.4426950408889634f * v);
    return __builtin_amdgcn_rcpf(1.0f + e);
}
__device__ __forceinline__ float fast_tanh(float v) {
    float t = __builtin_amdgcn_exp2f(2.8853900817779268f * v);   // exp(2v)
    return 1.0f - 2.0f * __builtin_amdgcn_rcpf(t + 1.0f);
}

// ---------------------------------------------------------------------------
// Pre-pass: f32 -> f16 conversion (x once; W_i|W_f|W_g|W_o concatenated).
// ---------------------------------------------------------------------------
__global__ __launch_bounds__(256)
void cvt_f32_to_f16(const float* __restrict__ src, _Float16* __restrict__ dst, int n)
{
    int i = (blockIdx.x * 256 + threadIdx.x) * 4;
    if (i + 3 < n) {
        float4 v = *(const float4*)(src + i);
        _Float16* d = dst + i;
        d[0] = (_Float16)v.x; d[1] = (_Float16)v.y;
        d[2] = (_Float16)v.z; d[3] = (_Float16)v.w;
    }
}

// ---------------------------------------------------------------------------
// Phase 1: gates[m, n] = act( x[m,:] . Wcat[n,:] + bcat[n] )
// M=65536, N=2048 (4 gates), K=512.  f16 WMMA, f32 accumulate.
// Double-buffered LDS, async global->LDS staging overlapped with WMMA.
// ---------------------------------------------------------------------------
__global__ __launch_bounds__(256)
void gates_gemm(const _Float16* __restrict__ x16,   // [M_,K_]
                const _Float16* __restrict__ w16,   // [N_,K_] gate-concatenated
                const float* __restrict__ bi, const float* __restrict__ bf,
                const float* __restrict__ bg, const float* __restrict__ bo,
                _Float16* __restrict__ gates)
{
    __shared__ _Float16 sx[2][TM * SXS];
    __shared__ _Float16 sw[2][TN * SWS];

    const int m0 = blockIdx.x * TM;
    const int n0 = blockIdx.y * TN;

    const int tid  = threadIdx.x;
    const int lane = tid & 31;
    const int wave = tid >> 5;
    const int wm   = wave & 1;                    // M sub-tile (2 x 32)
    const int wn   = wave >> 1;                   // N sub-tile (4 x 32)
    const int l16  = lane & 15;
    const int hi   = lane >> 4;

    // async staging: 16B chunks; x tile 64x32 halves = 256 chunks (1/thread),
    // w tile 128x32 halves = 512 chunks (2/thread).  LDS dsts all 16B-aligned.
    auto stage = [&](int buf, int k0) {
        {
            int c = tid, row = c >> 2, cc = (c & 3) * 8;
            cp_async_b128(&sx[buf][row * SXS + cc],
                          x16 + (size_t)(m0 + row) * K_ + k0 + cc);
        }
        #pragma unroll
        for (int u = 0; u < 2; ++u) {
            int c = tid * 2 + u, row = c >> 2, cc = (c & 3) * 8;
            cp_async_b128(&sw[buf][row * SWS + cc],
                          w16 + (size_t)(n0 + row) * K_ + k0 + cc);
        }
    };

    v8f acc[2][2] = {};

    stage(0, 0);
    for (int ks = 0; ks < NKS; ++ks) {
        wait_async0();
        __syncthreads();                        // tile ks visible to all waves
        if (ks + 1 < NKS) stage((ks + 1) & 1, (ks + 1) * TK);
        const int buf = ks & 1;

        // fragments per the CDNA5 16-bit A (16x32) / B (32x16) VGPR layouts
        v16h afr[2], bfr[2];
        #pragma unroll
        for (int i = 0; i < 2; ++i) {
            int row = wm * 32 + i * 16 + l16;
            const v8h lo = *(const v8h*)&sx[buf][row * SXS + 8 * hi];
            const v8h h8 = *(const v8h*)&sx[buf][row * SXS + 16 + 8 * hi];
            #pragma unroll
            for (int j = 0; j < 8; ++j) { afr[i][j] = lo[j]; afr[i][j + 8] = h8[j]; }
        }
        #pragma unroll
        for (int jn = 0; jn < 2; ++jn) {
            int col = wn * 32 + jn * 16 + l16;
            const v8h lo = *(const v8h*)&sw[buf][col * SWS + 8 * hi];
            const v8h h8 = *(const v8h*)&sw[buf][col * SWS + 16 + 8 * hi];
            #pragma unroll
            for (int j = 0; j < 8; ++j) { bfr[jn][j] = lo[j]; bfr[jn][j + 8] = h8[j]; }
        }
        #pragma unroll
        for (int i = 0; i < 2; ++i)
            #pragma unroll
            for (int jn = 0; jn < 2; ++jn)
                acc[i][jn] = __builtin_amdgcn_wmma_f32_16x16x32_f16(
                    false, afr[i], false, bfr[jn], (short)0, acc[i][jn], false, false);
    }

    // epilogue: bias + activation (sigmoid for i,f,o; tanh for g), store f16
    const int p = n0 >> 9;                        // gate id, uniform per block
    const float* bsel = (p == 0) ? bi : (p == 1) ? bf : (p == 2) ? bg : bo;
    #pragma unroll
    for (int jn = 0; jn < 2; ++jn) {
        int n = n0 + wn * 32 + jn * 16 + l16;
        float bias = bsel[n & (H_ - 1)];
        #pragma unroll
        for (int i = 0; i < 2; ++i) {
            #pragma unroll
            for (int r = 0; r < 8; ++r) {
                int m = m0 + wm * 32 + i * 16 + r + 8 * hi;
                float v = acc[i][jn][r] + bias;
                v = (p == 2) ? fast_tanh(v) : fast_sigmoid(v);
                gates[(size_t)m * N_ + n] = (_Float16)v;
            }
        }
    }
}

// ---------------------------------------------------------------------------
// Phase 2: elementwise scan over T; only c recurs, h needed only at t=T-1.
// ---------------------------------------------------------------------------
__global__ __launch_bounds__(256)
void lstm_scan(const _Float16* __restrict__ gates, float* __restrict__ hout)
{
    int idx = blockIdx.x * 256 + threadIdx.x;    // 0..16383
    int b   = idx >> 9;
    int hh  = idx & (H_ - 1);
    const _Float16* base = gates + (size_t)b * T_ * N_;
    float c = 0.0f;
    for (int t = 0; t < T_; ++t) {
        const _Float16* row = base + (size_t)t * N_;
        float gi = (float)row[hh];
        float gf = (float)row[H_ + hh];
        float gg = (float)row[2 * H_ + hh];
        c = gf * c + gi * gg;
    }
    float go = (float)(base + (size_t)(T_ - 1) * N_)[3 * H_ + hh];
    hout[idx] = go * fast_tanh(c);
}

// ---------------------------------------------------------------------------
// Phase 3: out[b, o] = h[b,:] . W_fc[o,:] + b_fc[o]   (32 x 256 x 512, tiny)
// ---------------------------------------------------------------------------
__global__ __launch_bounds__(256)
void fc_kernel(const float* __restrict__ hbuf, const float* __restrict__ Wfc,
               const float* __restrict__ bfc, float* __restrict__ out)
{
    int b = blockIdx.x;        // 0..31
    int o = threadIdx.x;       // 0..255
    const float* hr = hbuf + b * H_;
    const float* wr = Wfc  + o * H_;
    float s = 0.0f;
    #pragma unroll 4
    for (int h = 0; h < H_; h += 4) {
        float4 hv = *(const float4*)(hr + h);
        float4 wv = *(const float4*)(wr + h);
        s += hv.x * wv.x + hv.y * wv.y + hv.z * wv.z + hv.w * wv.w;
    }
    out[b * O_ + o] = s + bfc[o];
}

// ---------------------------------------------------------------------------
extern "C" void kernel_launch(void* const* d_in, const int* in_sizes, int n_in,
                              void* d_out, int out_size, void* d_ws, size_t ws_size,
                              hipStream_t stream)
{
    const float* x   = (const float*)d_in[0];
    const float* Wi  = (const float*)d_in[1];
    const float* bi  = (const float*)d_in[2];
    const float* Wf  = (const float*)d_in[3];
    const float* bf  = (const float*)d_in[4];
    const float* Wg  = (const float*)d_in[5];
    const float* bg  = (const float*)d_in[6];
    const float* Wo  = (const float*)d_in[7];
    const float* bo  = (const float*)d_in[8];
    const float* Wfc = (const float*)d_in[9];
    const float* bfc = (const float*)d_in[10];

    char* ws = (char*)d_ws;
    _Float16* gates = (_Float16*)ws;                                  // 256 MB
    _Float16* x16   = (_Float16*)(ws + (size_t)M_ * N_ * 2);          //  64 MB
    _Float16* w16   = (_Float16*)(ws + (size_t)M_ * N_ * 2 + (size_t)M_ * K_ * 2);
    float*    hbuf  = (float*)((char*)w16 + (size_t)N_ * K_ * 2);

    // pre-convert x and the four gate weight matrices to f16
    {
        int nx = M_ * K_;                         // 33,554,432
        cvt_f32_to_f16<<<nx / (4 * 256), 256, 0, stream>>>(x, x16, nx);
        int nw = H_ * K_;                         // 262,144 per gate
        cvt_f32_to_f16<<<nw / (4 * 256), 256, 0, stream>>>(Wi, w16 + 0 * (size_t)nw, nw);
        cvt_f32_to_f16<<<nw / (4 * 256), 256, 0, stream>>>(Wf, w16 + 1 * (size_t)nw, nw);
        cvt_f32_to_f16<<<nw / (4 * 256), 256, 0, stream>>>(Wg, w16 + 2 * (size_t)nw, nw);
        cvt_f32_to_f16<<<nw / (4 * 256), 256, 0, stream>>>(Wo, w16 + 3 * (size_t)nw, nw);
    }

    dim3 g1(M_ / TM, N_ / TN);   // 1024 x 16 blocks
    gates_gemm<<<g1, 256, 0, stream>>>(x16, w16, bi, bf, bg, bo, gates);
    lstm_scan<<<(B_ * H_) / 256, 256, 0, stream>>>(gates, hbuf);
    fc_kernel<<<B_, 256, 0, stream>>>(hbuf, Wfc, bfc, (float*)d_out);
}
